// NNTimeSeriesModel_78108275245099
// MI455X (gfx1250) — matrix operand order repaired and checked by reading, not compiled, not run on hardware
//
#include <hip/hip_runtime.h>

typedef _Float16 half8 __attribute__((ext_vector_type(8)));
typedef _Float16 v16h  __attribute__((ext_vector_type(16)));
typedef float    v8f   __attribute__((ext_vector_type(8)));

union V16U { v16h v; half8 h[2]; };

#define Bsz 1024
#define Tn  512
#define Hn  64
#define Gn  256   // 4*H
#define Ln  4

#define L2E 1.4426950408889634f

// Fused "my LDS ops retired -> signal" (single asm block so nothing drifts in
// between), split wait, and a hard scheduling fence to keep compute between
// signal and wait.
__device__ __forceinline__ void lds_fence_signal() {
    asm volatile("s_wait_dscnt 0x0\n\ts_barrier_signal -1" ::: "memory");
}
__device__ __forceinline__ void bar_wait() {
    asm volatile("s_barrier_wait -1" ::: "memory");
}
#define SCHED_FENCE() __builtin_amdgcn_sched_barrier(0)

// ---------------------------------------------------------------------------
// Prep: f32 weights -> f16 (already in WMMA-B friendly row-major [col][k]),
// bias sum, and x transposed to time-major [T][B] f32.
// ---------------------------------------------------------------------------
__global__ void prep_kernel(const float* __restrict__ x,
                            const float* __restrict__ w_ih_rest,
                            const float* __restrict__ w_hh,
                            const float* __restrict__ b_ih,
                            const float* __restrict__ b_hh,
                            _Float16* __restrict__ wh16,
                            _Float16* __restrict__ wi16,
                            float* __restrict__ bsum,
                            float* __restrict__ xT)
{
    const int NWH = Ln * Gn * Hn;        // 65536
    const int NWI = (Ln - 1) * Gn * Hn;  // 49152
    const int NB  = Ln * Gn;             // 1024
    const int NX  = Tn * Bsz;            // 524288
    const int total = NWH + NWI + NB + NX;
    for (int i = blockIdx.x * blockDim.x + threadIdx.x; i < total;
         i += gridDim.x * blockDim.x) {
        int j = i;
        if (j < NWH) { wh16[j] = (_Float16)w_hh[j]; continue; }
        j -= NWH;
        if (j < NWI) { wi16[j] = (_Float16)w_ih_rest[j]; continue; }
        j -= NWI;
        if (j < NB)  { bsum[j] = b_ih[j] + b_hh[j]; continue; }
        j -= NB;
        int t = j / Bsz, b = j - t * Bsz;          // xT[t][b] = x[b][t][0]
        xT[j] = x[(size_t)b * Tn + t];
    }
}

// ---------------------------------------------------------------------------
// One LSTM layer. Grid: B/32 blocks; block: 256 threads = 8 waves (2 M x 4 N).
// Each wave: M-tile of 16 batch rows, 16 hidden units, all 4 gates.
// Weights resident in VGPRs, h in LDS, c in VGPRs. 16 WMMAs/wave/step.
// Two rotated/overlapped split barriers per step:
//   W(t-1): waited at loop top (h_{t-1} committed); round-trip hides behind
//           the previous iteration's global h stores / x prefetch / latch.
//   R(t)  : signaled right after this wave's LDS reads retire (compute hides
//           the round-trip; sched_barrier pins the wait after compute),
//           waited just before the h_t scatter.
// h_t goes to the sequence buffer directly from registers (b16 stores).
// ---------------------------------------------------------------------------
template <bool LAYER0, bool LAST>
__global__ __launch_bounds__(256) void lstm_layer_kernel(
    const _Float16* __restrict__ seq_in,   // [T][B][H] f16 (layers >= 1)
    _Float16* __restrict__ seq_out,        // [T][B][H] f16
    const _Float16* __restrict__ whh,      // [256][64] f16 (this layer)
    const _Float16* __restrict__ wih,      // [256][64] f16 (layers >= 1)
    const float* __restrict__ w0,          // [256] f32 w_ih0 (layer 0, IN=1)
    const float* __restrict__ bsum,        // [256] f32 b_ih+b_hh (this layer)
    const float* __restrict__ xT)          // [T][B] f32 (layer 0)
{
    __shared__ __align__(16) _Float16 sh_h[32 * 72]; // 32 rows, stride 72 halves
    __shared__ float sh_x[2][32];

    const int tid  = threadIdx.x;
    const int lane = tid & 31;
    const int wave = tid >> 5;
    const int mw = wave >> 2;     // 0..1 : 16-row M tile
    const int nw = wave & 3;      // 0..3 : 16-unit N chunk
    const int lr = lane & 15;
    const int hi = lane >> 4;
    const int rowBase = blockIdx.x * 32;

    // --- Load weight B-tiles into registers (ISA B layout: lane=col, K halves
    //     in table order: 8 at hi*8, 8 at 16+hi*8) ---
    v16h bh[4][2], bi[4][2];
    float sb[4], w0g[4];
    const float sA = -L2E;         // sigmoid gates (i,f,o)
    const float sG = -2.f * L2E;   // tanh gate (g)
#pragma unroll
    for (int g = 0; g < 4; ++g) {
#pragma unroll
        for (int s = 0; s < 2; ++s) {
            const _Float16* wp = whh + (g * 64 + nw * 16 + lr) * 64 + s * 32 + hi * 16;
            { V16U u; u.h[0] = *(const half8*)(wp - hi * 8); u.h[1] = *(const half8*)(wp - hi * 8 + 16); bh[g][s] = u.v; }
            if (!LAYER0) {
                const _Float16* ip = wih + (g * 64 + nw * 16 + lr) * 64 + s * 32 + hi * 16;
                { V16U u; u.h[0] = *(const half8*)(ip - hi * 8); u.h[1] = *(const half8*)(ip - hi * 8 + 16); bi[g][s] = u.v; }
            }
        }
        float b = bsum[g * 64 + nw * 16 + lr];
        sb[g] = (g == 2 ? sG : sA) * b;   // bias folded into exp2 argument
        if (LAYER0) w0g[g] = w0[g * 64 + nw * 16 + lr];
    }

    // --- init h (LDS) and c (regs) to zero; stage x(0), prefetch x(1) ---
    for (int i = tid; i < 32 * 72; i += 256) sh_h[i] = (_Float16)0.f;
    v8f c;
#pragma unroll
    for (int r = 0; r < 8; ++r) c[r] = 0.f;

    float xreg = 0.f;
    if (LAYER0 && tid < 32) {
        sh_x[0][tid] = xT[rowBase + tid];              // x(0)
        xreg = xT[(size_t)Bsz + rowBase + tid];        // x(1)
    }

    // software pipelining: prefetch x tile for t=0 (layers >= 1)
    V16U axc0, axc1;
    if (!LAYER0) {
        const _Float16* xr = seq_in + ((size_t)rowBase + mw * 16 + lr) * 64;
        axc0.h[0] = *(const half8*)&xr[hi * 8];
        axc0.h[1] = *(const half8*)&xr[16 + hi * 8];
        axc1.h[0] = *(const half8*)&xr[32 + hi * 8];
        axc1.h[1] = *(const half8*)&xr[48 + hi * 8];
    }

    const _Float16* hp = &sh_h[(mw * 16 + lr) * 72];
    const int sIdx = (mw * 16 + hi * 8) * 72 + nw * 16 + lr;  // scatter base

    lds_fence_signal();   // W(-1): init writes committed

#pragma unroll 2
    for (int t = 0; t < Tn; ++t) {
        bar_wait();       // W(t-1): h_{t-1} (and staged x) visible in LDS

        // ---- read phase: A tiles of h_{t-1} (+ layer0 x scalars) ----
        V16U ah0, ah1;
        ah0.h[0] = *(const half8*)&hp[hi * 8];
        ah0.h[1] = *(const half8*)&hp[16 + hi * 8];
        ah1.h[0] = *(const half8*)&hp[32 + hi * 8];
        ah1.h[1] = *(const half8*)&hp[48 + hi * 8];
        float xv[8];
        if (LAYER0) {
#pragma unroll
            for (int r = 0; r < 8; ++r)
                xv[r] = sh_x[t & 1][mw * 16 + hi * 8 + r];
        }
        lds_fence_signal();   // R(t): this wave's LDS reads retired

        // prefetch x tile for t+1 (global; sits inside the barrier window)
        const int tn = (t + 1 < Tn) ? t + 1 : t;
        V16U axn0, axn1;
        if (!LAYER0) {
            const _Float16* xr =
                seq_in + ((size_t)tn * Bsz + rowBase + mw * 16 + lr) * 64;
            axn0.h[0] = *(const half8*)&xr[hi * 8];
            axn0.h[1] = *(const half8*)&xr[16 + hi * 8];
            axn1.h[0] = *(const half8*)&xr[32 + hi * 8];
            axn1.h[1] = *(const half8*)&xr[48 + hi * 8];
        }

        // ---- compute: 16 WMMAs + gate math (TRANS32 co-executes) ----
        v8f acc[4];
#pragma unroll
        for (int g = 0; g < 4; ++g) {
            v8f a = {};   // first WMMA uses inline SRC2 = 0
            a = __builtin_amdgcn_wmma_f32_16x16x32_f16(false, ah0.v, false, bh[g][0],
                                                       (short)0, a, false, false);
            a = __builtin_amdgcn_wmma_f32_16x16x32_f16(false, ah1.v, false, bh[g][1],
                                                       (short)0, a, false, false);
            if (!LAYER0) {
                a = __builtin_amdgcn_wmma_f32_16x16x32_f16(false, axc0.v, false, bi[g][0],
                                                           (short)0, a, false, false);
                a = __builtin_amdgcn_wmma_f32_16x16x32_f16(false, axc1.v, false, bi[g][1],
                                                           (short)0, a, false, false);
            }
            acc[g] = a;
        }

        if (LAYER0) {
            // rank-1 input projection: gates += x[t,b] * w_ih0[col]
#pragma unroll
            for (int r = 0; r < 8; ++r)
#pragma unroll
                for (int g = 0; g < 4; ++g)
                    acc[g][r] = fmaf(xv[r], w0g[g], acc[g][r]);
        }

        _Float16 hval[8];
#pragma unroll
        for (int r = 0; r < 8; ++r) {
            float ei = __builtin_amdgcn_exp2f(fmaf(acc[0][r], sA, sb[0]));
            float ef = __builtin_amdgcn_exp2f(fmaf(acc[1][r], sA, sb[1]));
            float eg = __builtin_amdgcn_exp2f(fmaf(acc[2][r], sG, sb[2]));
            float eo = __builtin_amdgcn_exp2f(fmaf(acc[3][r], sA, sb[3]));
            float iv = __builtin_amdgcn_rcpf(1.f + ei);
            float fv = __builtin_amdgcn_rcpf(1.f + ef);
            float gv = fmaf(2.f, __builtin_amdgcn_rcpf(1.f + eg), -1.f);
            float ov = __builtin_amdgcn_rcpf(1.f + eo);
            float cv = fmaf(fv, c[r], iv * gv);
            c[r] = cv;
            float ec = __builtin_amdgcn_exp2f(cv * sG);
            float tv = fmaf(2.f, __builtin_amdgcn_rcpf(1.f + ec), -1.f);
            hval[r] = (_Float16)(ov * tv);
        }

        SCHED_FENCE();   // keep the compute above between signal and wait
        bar_wait();      // R(t) complete: safe to overwrite sh_h / sh_x

        // ---- write phase: scatter h_t (+ stage x(t+1) for layer 0) ----
#pragma unroll
        for (int r = 0; r < 8; ++r)
            sh_h[sIdx + r * 72] = hval[r];
        if (LAYER0) {
            if (tid < 32) sh_x[(t + 1) & 1][tid] = xreg;   // x(t+1)
        }
        lds_fence_signal();   // W(t): h_t (and x stage) committed

        // ---- overlap window: global h stores + prefetches hide W(t) ----
        if (!LAST || t == Tn - 1) {
            _Float16* gp = seq_out +
                ((size_t)t * Bsz + rowBase + mw * 16 + hi * 8) * 64 + nw * 16 + lr;
#pragma unroll
            for (int r = 0; r < 8; ++r) gp[r * 64] = hval[r];
        }
        if (LAYER0) {
            int t2 = (t + 2 < Tn) ? t + 2 : Tn - 1;
            if (tid < 32) xreg = xT[(size_t)t2 * Bsz + rowBase + tid];
        }
        if (!LAYER0) { axc0 = axn0; axc1 = axn1; }
    }
}

// ---------------------------------------------------------------------------
// FC head: out[b] = fc2_b + sum_j relu(h_last[b] . fc1_w[j] + fc1_b[j]) * fc2_w[j]
// ---------------------------------------------------------------------------
__global__ __launch_bounds__(128) void fc_kernel(
    const _Float16* __restrict__ seq,   // layer-3 output, [T][B][64]
    const float* __restrict__ fc1_w,    // [128][64]
    const float* __restrict__ fc1_b,    // [128]
    const float* __restrict__ fc2_w,    // [1][128]
    const float* __restrict__ fc2_b,    // [1]
    float* __restrict__ out)            // [B][1]
{
    __shared__ float red[128];
    const int b = blockIdx.x;
    const int j = threadIdx.x;
    const _Float16* h = seq + ((size_t)(Tn - 1) * Bsz + b) * 64;
    float s = 0.f;
#pragma unroll
    for (int k = 0; k < 64; ++k) s += (float)h[k] * fc1_w[j * 64 + k];
    s += fc1_b[j];
    s = fmaxf(s, 0.f);
    red[j] = s * fc2_w[j];
    __syncthreads();
    for (int off = 64; off > 0; off >>= 1) {
        if (j < off) red[j] += red[j + off];
        __syncthreads();
    }
    if (j == 0) out[b] = red[0] + fc2_b[0];
}

// ---------------------------------------------------------------------------
extern "C" void kernel_launch(void* const* d_in, const int* in_sizes, int n_in,
                              void* d_out, int out_size, void* d_ws, size_t ws_size,
                              hipStream_t stream) {
    (void)in_sizes; (void)n_in; (void)out_size; (void)ws_size;
    const float* x         = (const float*)d_in[0];
    const float* w_ih0     = (const float*)d_in[1];
    const float* w_ih_rest = (const float*)d_in[2];
    const float* w_hh      = (const float*)d_in[3];
    const float* b_ih      = (const float*)d_in[4];
    const float* b_hh      = (const float*)d_in[5];
    const float* fc1_w     = (const float*)d_in[6];
    const float* fc1_b     = (const float*)d_in[7];
    const float* fc2_w     = (const float*)d_in[8];
    const float* fc2_b     = (const float*)d_in[9];
    float* out = (float*)d_out;

    char* ws = (char*)d_ws;
    size_t off = 0;
    _Float16* seqA = (_Float16*)(ws + off); off += (size_t)Tn * Bsz * Hn * 2;
    _Float16* seqB = (_Float16*)(ws + off); off += (size_t)Tn * Bsz * Hn * 2;
    _Float16* wh16 = (_Float16*)(ws + off); off += (size_t)Ln * Gn * Hn * 2;
    _Float16* wi16 = (_Float16*)(ws + off); off += (size_t)(Ln - 1) * Gn * Hn * 2;
    float* bsum    = (float*)(ws + off);    off += (size_t)Ln * Gn * 4;
    float* xT      = (float*)(ws + off);    off += (size_t)Tn * Bsz * 4;

    prep_kernel<<<512, 256, 0, stream>>>(x, w_ih_rest, w_hh, b_ih, b_hh,
                                         wh16, wi16, bsum, xT);

    // layer 0 (input via rank-1 x * w_ih0)
    lstm_layer_kernel<true, false><<<Bsz / 32, 256, 0, stream>>>(
        nullptr, seqA, wh16, nullptr, w_ih0, bsum, xT);

    // layers 1..2 ping-pong; layer 3 only stores t = T-1
    lstm_layer_kernel<false, false><<<Bsz / 32, 256, 0, stream>>>(
        seqA, seqB, wh16 + 1 * Gn * Hn, wi16 + 0 * Gn * Hn, nullptr,
        bsum + 1 * Gn, nullptr);
    lstm_layer_kernel<false, false><<<Bsz / 32, 256, 0, stream>>>(
        seqB, seqA, wh16 + 2 * Gn * Hn, wi16 + 1 * Gn * Hn, nullptr,
        bsum + 2 * Gn, nullptr);
    lstm_layer_kernel<false, true><<<Bsz / 32, 256, 0, stream>>>(
        seqA, seqB, wh16 + 3 * Gn * Hn, wi16 + 2 * Gn * Hn, nullptr,
        bsum + 3 * Gn, nullptr);

    fc_kernel<<<Bsz, 128, 0, stream>>>(seqB, fc1_w, fc1_b, fc2_w, fc2_b, out);
}
